// TransformerPolicy_4844723109926
// MI455X (gfx1250) — matrix operand order repaired and checked by reading, not compiled, not run on hardware
//
#include <hip/hip_runtime.h>
#include <hip/hip_bf16.h>
#include <math.h>

// ---------- types ----------
typedef __bf16 bfloat;
typedef __attribute__((ext_vector_type(16))) __bf16 v16bf;
typedef __attribute__((ext_vector_type(8)))  __bf16 v8bf;
typedef __attribute__((ext_vector_type(8)))  float  v8f;
typedef int v4i __attribute__((vector_size(16)));
typedef __attribute__((address_space(1))) v4i as1v4i;   // global int4 (matches builtin param)
typedef __attribute__((address_space(3))) v4i as3v4i;   // LDS int4
typedef __attribute__((address_space(1))) void as1void;
typedef __attribute__((address_space(3))) void as3void;

// ---------- model dims ----------
#define BSZ   16
#define Hh    24
#define Ww    24
#define Ntok  576          // H*W
#define Gtok  2
#define Ttok  579          // N + 1 + G
#define Dm    256
#define NH    8
#define HD    32
#define DEPTH 6
#define MLPD  512
#define CIN   10
#define BINS  82
#define TP    608          // padded T for attention buffers (mult of 32)
#define KCHUNKS 19         // ceil(579/32)
#define QTILES  37         // ceil(579/16)

#if defined(__AMDGCN__) && __has_builtin(__builtin_amdgcn_global_load_async_to_lds_b128)
#define HAVE_ASYNC_LDS 1
#else
#define HAVE_ASYNC_LDS 0
#endif

__device__ inline void wait_async0() {
#if defined(__AMDGCN__) && __has_builtin(__builtin_amdgcn_s_wait_asynccnt)
  __builtin_amdgcn_s_wait_asynccnt(0);
#else
  asm volatile("s_wait_asynccnt 0x0" ::: "memory");
#endif
}

__device__ inline v8f wmma_bf16(v16bf a, v16bf b, v8f c) {
  return __builtin_amdgcn_wmma_f32_16x16x32_bf16(false, a, false, b, (short)0, c,
                                                 false, false);
}

__device__ inline float gelu_exact(float x) {
  return 0.5f * x * (1.0f + erff(x * 0.70710678118654752f));
}

// ---------- weight transpose+convert: f32 [K][N] -> bf16 [N][K] ----------
__global__ __launch_bounds__(256)
void wconv_kernel(const float* __restrict__ w, bfloat* __restrict__ wt, int K, int N) {
  int i = blockIdx.x * 256 + threadIdx.x;
  if (i >= K * N) return;
  int k = i / N, n = i % N;
  wt[(size_t)n * K + k] = (bfloat)w[i];
}

// ---------- embedding: obs @ input_w + bias + pos, cls/global tokens ----------
__global__ __launch_bounds__(256)
void embed_kernel(const float* __restrict__ obs, const float* __restrict__ iw,
                  const float* __restrict__ ib, const float* __restrict__ cls,
                  const float* __restrict__ glob, const float* __restrict__ rowe,
                  const float* __restrict__ cole, float* __restrict__ X) {
  int row = blockIdx.x;              // b*T + t
  int d = threadIdx.x;
  int b = row / Ttok, t = row % Ttok;
  float v;
  if (t == 0) {
    v = cls[d];
  } else if (t <= Gtok) {
    v = glob[(t - 1) * Dm + d];
  } else {
    int n = t - 1 - Gtok;
    float s = ib[d];
#pragma unroll
    for (int c = 0; c < CIN; c++)
      s += obs[((size_t)b * CIN + c) * Ntok + n] * iw[c * Dm + d];
    v = s + rowe[(n / Ww) * Dm + d] + cole[(n % Ww) * Dm + d];
  }
  X[(size_t)row * Dm + d] = v;
}

// ---------- layernorm (wave per row of 256), optional bf16 out ----------
// MAP: 0 dense rows; 1 policy rows (x[:, 1:N+1] slice); 2 value rows (cls only)
template <int MAP, bool OUTBF>
__global__ __launch_bounds__(256)
void ln_kernel(const float* __restrict__ X, const float* __restrict__ g,
               const float* __restrict__ bta, void* __restrict__ out, int rows) {
  int r = blockIdx.x * 8 + (threadIdx.x >> 5);
  int lane = threadIdx.x & 31;
  if (r >= rows) return;
  long srcRow;
  if (MAP == 0)      srcRow = r;
  else if (MAP == 1) srcRow = (long)(r / Ntok) * Ttok + 1 + (r % Ntok); // b*T + 1 + n
  else               srcRow = (long)r * Ttok;                          // b*T (cls)
  const float* x = X + srcRow * Dm;
  float v[8];
  float s = 0.f;
#pragma unroll
  for (int i = 0; i < 8; i++) { v[i] = x[lane + i * 32]; s += v[i]; }
#pragma unroll
  for (int m = 16; m; m >>= 1) s += __shfl_xor(s, m, 32);
  float mean = s * (1.0f / 256.0f);
  float vs = 0.f;
#pragma unroll
  for (int i = 0; i < 8; i++) { v[i] -= mean; vs += v[i] * v[i]; }
#pragma unroll
  for (int m = 16; m; m >>= 1) vs += __shfl_xor(vs, m, 32);
  float rstd = rsqrtf(vs * (1.0f / 256.0f) + 1e-5f);
#pragma unroll
  for (int i = 0; i < 8; i++) {
    int c = lane + i * 32;
    float o = v[i] * rstd * g[c] + bta[c];
    if (OUTBF) ((bfloat*)out)[(size_t)r * Dm + c] = (bfloat)o;
    else       ((float*)out)[(size_t)r * Dm + c] = o;
  }
}

// ---------- GEMM: C[M,N] = A(bf16,[M][K]) * Wt(bf16,[N][K]) + bias ----------
// 256 threads = 8 waves; block tile 16(M) x 128(N); wave per 16x16 via WMMA.
// OUTMODE: 0 = f32 [M][N] (opt residual), 1 = bf16 [M][N], 2 = qkv split
template <int K, bool RES, bool GELU_ACT, int OUTMODE>
__global__ __launch_bounds__(256)
void gemm_kernel(const bfloat* __restrict__ A, const bfloat* __restrict__ Wt,
                 const float* __restrict__ bias, const float* __restrict__ res,
                 void* __restrict__ outp, int M, int N) {
  __shared__ bfloat As[16 * K];
  int tid = threadIdx.x;
  int wave = tid >> 5, lane = tid & 31;
  int m0 = blockIdx.x * 16;
  int n0 = blockIdx.y * 128 + wave * 16;
  int half = lane >> 4, nl = lane & 15;

  // prefetch this wave's weight panel start (global_prefetch_b8)
  __builtin_prefetch(&Wt[(size_t)(n0 + nl) * K], 0, 3);

  // stage A tile (16 x K bf16) into LDS
  const int chunks = (16 * K) >> 3;  // 16-byte chunks
#if HAVE_ASYNC_LDS
  for (int c = tid; c < chunks; c += 256) {
    int e = c << 3;
    int row = e / K, col = e % K;
    __builtin_amdgcn_global_load_async_to_lds_b128(
        (as1v4i*)(as1void*)(void*)&A[(size_t)(m0 + row) * K + col],
        (as3v4i*)(as3void*)&As[row * K + col], 0, 0);
  }
  wait_async0();
  __syncthreads();
#else
  for (int c = tid; c < chunks; c += 256) {
    int e = c << 3;
    int row = e / K, col = e % K;
    *(v8bf*)&As[row * K + col] = *(const v8bf*)&A[(size_t)(m0 + row) * K + col];
  }
  __syncthreads();
#endif

  v8f acc = {0.f, 0.f, 0.f, 0.f, 0.f, 0.f, 0.f, 0.f};
#pragma unroll 4
  for (int k0 = 0; k0 < K; k0 += 32) {
    v16bf af, bf_;
    *(v8bf*)&af        = *(const v8bf*)&As[nl * K + k0 + half * 16];
    *((v8bf*)&af + 1)  = *(const v8bf*)&As[nl * K + k0 + half * 16 + 8];
    bf_ = *(const v16bf*)&Wt[(size_t)(n0 + nl) * K + k0 + half * 16];
    acc = wmma_bf16(af, bf_, acc);
  }

  int n = n0 + nl;
  float bv = bias[n];
#pragma unroll
  for (int r = 0; r < 8; r++) {
    int m = m0 + r + half * 8;
    float v = acc[r] + bv;
    if (GELU_ACT) v = gelu_exact(v);
    if (RES) v += res[(size_t)m * N + n];
    if (OUTMODE == 0) {
      ((float*)outp)[(size_t)m * N + n] = v;
    } else if (OUTMODE == 1) {
      ((bfloat*)outp)[(size_t)m * N + n] = (bfloat)v;
    } else {
      // split into q/k/v bf16 buffers; q,k: [b*8+h][t][32]; v transposed [b*8+h][d][TP]
      int bb = m / Ttok, t = m % Ttok;
      int which = n >> 8, h = (n >> 5) & 7, d = n & 31;
      bfloat* base = (bfloat*)outp + (size_t)which * ((size_t)BSZ * NH * TP * HD);
      size_t bh = (size_t)(bb * NH + h);
      if (which < 2) base[(bh * TP + t) * HD + d] = (bfloat)v;
      else           base[(bh * HD + d) * TP + t] = (bfloat)v;
    }
  }
}

// ---------- attention: one wave per (b, head, 16-query tile), flash style ----------
__global__ __launch_bounds__(32)
void attn_kernel(const bfloat* __restrict__ qb, const bfloat* __restrict__ kbuf,
                 const bfloat* __restrict__ vtb, const float* __restrict__ rbias,
                 const int* __restrict__ ridx, bfloat* __restrict__ ob) {
  __shared__ bfloat Pl[16 * 32];
  __shared__ float bl[96];
  int lane = threadIdx.x & 31;
  int qt = blockIdx.x, h = blockIdx.y, b = blockIdx.z;
  for (int i = lane; i < BINS; i += 32) bl[i] = rbias[h * BINS + i];
  __syncthreads();

  int half = lane >> 4, nl = lane & 15;
  size_t bh = (size_t)(b * NH + h);
  int q0 = qt * 16;
  const bfloat* qrow = qb + (bh * TP + q0 + nl) * HD;
  v16bf qf;
  *(v8bf*)&qf       = *(const v8bf*)(qrow + half * 16);
  *((v8bf*)&qf + 1) = *(const v8bf*)(qrow + half * 16 + 8);

  float mrow[8], lrow[8];
  v8f acc0 = {0.f, 0.f, 0.f, 0.f, 0.f, 0.f, 0.f, 0.f};
  v8f acc1 = {0.f, 0.f, 0.f, 0.f, 0.f, 0.f, 0.f, 0.f};
#pragma unroll
  for (int r = 0; r < 8; r++) { mrow[r] = -1e30f; lrow[r] = 0.f; }
  const float scale = 0.17677669529663687f;  // 1/sqrt(HD)

  for (int jc = 0; jc < KCHUNKS; jc++) {
    int kb0 = jc * 32;
    v16bf kf0 = *(const v16bf*)(kbuf + (bh * TP + kb0 + nl) * HD + half * 16);
    v16bf kf1 = *(const v16bf*)(kbuf + (bh * TP + kb0 + 16 + nl) * HD + half * 16);
    v8f z = {0.f, 0.f, 0.f, 0.f, 0.f, 0.f, 0.f, 0.f};
    v8f s0 = wmma_bf16(qf, kf0, z);
    v8f s1 = wmma_bf16(qf, kf1, z);

    int key0 = kb0 + nl, key1 = key0 + 16;
    float pr0[8], pr1[8];
#pragma unroll
    for (int r = 0; r < 8; r++) {
      int mq = q0 + r + half * 8;
      if (mq > Ttok - 1) mq = Ttok - 1;
      float f0 = (key0 < Ttok) ? s0[r] * scale + bl[ridx[mq * Ttok + key0]] : -1e30f;
      float f1 = (key1 < Ttok) ? s1[r] * scale + bl[ridx[mq * Ttok + key1]] : -1e30f;
      float rm = fmaxf(f0, f1);
#pragma unroll
      for (int msk = 8; msk; msk >>= 1) rm = fmaxf(rm, __shfl_xor(rm, msk, 32));
      float mnew = fmaxf(mrow[r], rm);
      float corr = __expf(mrow[r] - mnew);
      float p0 = __expf(f0 - mnew);
      float p1 = __expf(f1 - mnew);
      float rs = p0 + p1;
#pragma unroll
      for (int msk = 8; msk; msk >>= 1) rs += __shfl_xor(rs, msk, 32);
      lrow[r] = lrow[r] * corr + rs;
      acc0[r] *= corr;
      acc1[r] *= corr;
      mrow[r] = mnew;
      pr0[r] = p0; pr1[r] = p1;
    }
    // C-layout -> LDS (row-major 16x32) -> A-layout fragment
#pragma unroll
    for (int r = 0; r < 8; r++) {
      Pl[(r + half * 8) * 32 + nl]      = (bfloat)pr0[r];
      Pl[(r + half * 8) * 32 + 16 + nl] = (bfloat)pr1[r];
    }
    __syncthreads();
    v16bf pf;
    *(v8bf*)&pf       = *(const v8bf*)&Pl[nl * 32 + half * 16];
    *((v8bf*)&pf + 1) = *(const v8bf*)&Pl[nl * 32 + half * 16 + 8];
    v16bf vf0 = *(const v16bf*)(vtb + (bh * HD + nl) * TP + kb0 + half * 16);
    v16bf vf1 = *(const v16bf*)(vtb + (bh * HD + 16 + nl) * TP + kb0 + half * 16);
    acc0 = wmma_bf16(pf, vf0, acc0);
    acc1 = wmma_bf16(pf, vf1, acc1);
    __syncthreads();
  }
#pragma unroll
  for (int r = 0; r < 8; r++) {
    int mq = q0 + r + half * 8;
    if (mq < Ttok) {
      float inv = 1.0f / lrow[r];
      size_t row = ((size_t)b * Ttok + mq) * Dm + h * HD;
      ob[row + nl]      = (bfloat)(acc0[r] * inv);
      ob[row + 16 + nl] = (bfloat)(acc1[r] * inv);
    }
  }
}

// ---------- head final dot: out = A(bf16, rows x 256) . w + b2 ----------
template <bool IS_VAL>
__global__ __launch_bounds__(256)
void head_dot_kernel(const bfloat* __restrict__ A, const float* __restrict__ w,
                     const float* __restrict__ b2, float* __restrict__ out, int rows) {
  int r = blockIdx.x * 8 + (threadIdx.x >> 5);
  int lane = threadIdx.x & 31;
  if (r >= rows) return;
  const bfloat* a = A + (size_t)r * Dm;
  float s = 0.f;
#pragma unroll
  for (int i = 0; i < 8; i++) s += (float)a[lane + i * 32] * w[lane + i * 32];
#pragma unroll
  for (int m = 16; m; m >>= 1) s += __shfl_xor(s, m, 32);
  if (lane == 0) {
    float v = s + b2[0];
    int oi = IS_VAL ? (r * (Ntok + 1) + Ntok)
                    : ((r / Ntok) * (Ntok + 1) + (r % Ntok));
    out[oi] = v;
  }
}

// ---------- host orchestration ----------
extern "C" void kernel_launch(void* const* d_in, const int* in_sizes, int n_in,
                              void* d_out, int out_size, void* d_ws, size_t ws_size,
                              hipStream_t stream) {
  (void)in_sizes; (void)n_in; (void)out_size; (void)ws_size;
  const float* obs      = (const float*)d_in[0];
  const float* input_w  = (const float*)d_in[1];
  const float* input_b  = (const float*)d_in[2];
  const float* cls_tok  = (const float*)d_in[3];
  const float* glob_tok = (const float*)d_in[4];
  const float* row_emb  = (const float*)d_in[5];
  const float* col_emb  = (const float*)d_in[6];
  const float* ln1_g    = (const float*)d_in[7];
  const float* ln1_b    = (const float*)d_in[8];
  const float* qkv_w    = (const float*)d_in[9];
  const float* qkv_b    = (const float*)d_in[10];
  const float* proj_w   = (const float*)d_in[11];
  const float* proj_b   = (const float*)d_in[12];
  const float* rel_bias = (const float*)d_in[13];
  const float* ln2_g    = (const float*)d_in[14];
  const float* ln2_b    = (const float*)d_in[15];
  const float* fc1_w    = (const float*)d_in[16];
  const float* fc1_b    = (const float*)d_in[17];
  const float* fc2_w    = (const float*)d_in[18];
  const float* fc2_b    = (const float*)d_in[19];
  const float* norm_g   = (const float*)d_in[20];
  const float* norm_b   = (const float*)d_in[21];
  const float* pol_ln_g = (const float*)d_in[22];
  const float* pol_ln_b = (const float*)d_in[23];
  const float* pol_w1   = (const float*)d_in[24];
  const float* pol_b1   = (const float*)d_in[25];
  const float* pol_w2   = (const float*)d_in[26];
  const float* pol_b2   = (const float*)d_in[27];
  const float* val_ln_g = (const float*)d_in[28];
  const float* val_ln_b = (const float*)d_in[29];
  const float* val_w1   = (const float*)d_in[30];
  const float* val_b1   = (const float*)d_in[31];
  const float* val_w2   = (const float*)d_in[32];
  const float* val_b2   = (const float*)d_in[33];
  const int*   rel_idx  = (const int*)d_in[34];
  float* out = (float*)d_out;

  const size_t ROWS = (size_t)BSZ * Ttok;             // 9264
  const size_t QKVS = (size_t)BSZ * NH * TP * HD;     // elems per q/k/v tensor

  char* W = (char*)d_ws;
  size_t off = 0;
  auto nxt = [&](size_t bytes) { size_t o = off; off += (bytes + 255) & ~(size_t)255; return o; };
  size_t o_wqkv = nxt((size_t)DEPTH * 768 * 256 * 2);
  size_t o_wprj = nxt((size_t)DEPTH * 256 * 256 * 2);
  size_t o_wfc1 = nxt((size_t)DEPTH * 512 * 256 * 2);
  size_t o_wfc2 = nxt((size_t)DEPTH * 256 * 512 * 2);
  size_t o_wpol = nxt((size_t)256 * 256 * 2);
  size_t o_wval = nxt((size_t)256 * 256 * 2);
  size_t o_X    = nxt(ROWS * Dm * 4);
  size_t o_H    = nxt(ROWS * Dm * 2);
  size_t o_QKV  = nxt(3 * QKVS * 2);                  // also reused as Y (f32) later
  size_t o_F    = nxt(ROWS * MLPD * 2);
  size_t o_VH   = nxt((size_t)BSZ * Dm * 2);
  size_t o_VF   = nxt((size_t)BSZ * Dm * 2);

  bfloat* WQKV = (bfloat*)(W + o_wqkv);
  bfloat* WPRJ = (bfloat*)(W + o_wprj);
  bfloat* WFC1 = (bfloat*)(W + o_wfc1);
  bfloat* WFC2 = (bfloat*)(W + o_wfc2);
  bfloat* WPOL = (bfloat*)(W + o_wpol);
  bfloat* WVAL = (bfloat*)(W + o_wval);
  float*  X    = (float*)(W + o_X);
  bfloat* Hb   = (bfloat*)(W + o_H);
  bfloat* QKVb = (bfloat*)(W + o_QKV);
  float*  Y    = (float*)(W + o_QKV);
  bfloat* Fb   = (bfloat*)(W + o_F);
  bfloat* VHb  = (bfloat*)(W + o_VH);
  bfloat* VFb  = (bfloat*)(W + o_VF);

  // weight transpose/convert to bf16 [N][K]
  for (int l = 0; l < DEPTH; l++) {
    wconv_kernel<<<(256 * 768 + 255) / 256, 256, 0, stream>>>(
        qkv_w + (size_t)l * 256 * 768, WQKV + (size_t)l * 768 * 256, 256, 768);
    wconv_kernel<<<(256 * 256 + 255) / 256, 256, 0, stream>>>(
        proj_w + (size_t)l * 256 * 256, WPRJ + (size_t)l * 256 * 256, 256, 256);
    wconv_kernel<<<(256 * 512 + 255) / 256, 256, 0, stream>>>(
        fc1_w + (size_t)l * 256 * 512, WFC1 + (size_t)l * 512 * 256, 256, 512);
    wconv_kernel<<<(512 * 256 + 255) / 256, 256, 0, stream>>>(
        fc2_w + (size_t)l * 512 * 256, WFC2 + (size_t)l * 256 * 512, 512, 256);
  }
  wconv_kernel<<<(256 * 256 + 255) / 256, 256, 0, stream>>>(pol_w1, WPOL, 256, 256);
  wconv_kernel<<<(256 * 256 + 255) / 256, 256, 0, stream>>>(val_w1, WVAL, 256, 256);

  // zero q/k/v padding regions once (deterministic each call)
  (void)hipMemsetAsync(QKVb, 0, 3 * QKVS * 2, stream);

  // embedding
  embed_kernel<<<(int)ROWS, 256, 0, stream>>>(obs, input_w, input_b, cls_tok,
                                              glob_tok, row_emb, col_emb, X);

  // transformer layers
  for (int l = 0; l < DEPTH; l++) {
    ln_kernel<0, true><<<(int)(ROWS / 8), 256, 0, stream>>>(
        X, ln1_g + l * Dm, ln1_b + l * Dm, Hb, (int)ROWS);
    gemm_kernel<256, false, false, 2><<<dim3(Ttok, 6), 256, 0, stream>>>(
        Hb, WQKV + (size_t)l * 768 * 256, qkv_b + l * 768, nullptr, QKVb,
        (int)ROWS, 768);
    attn_kernel<<<dim3(QTILES, NH, BSZ), 32, 0, stream>>>(
        QKVb, QKVb + QKVS, QKVb + 2 * QKVS, rel_bias + (size_t)l * NH * BINS,
        rel_idx, Hb);
    gemm_kernel<256, true, false, 0><<<dim3(Ttok, 2), 256, 0, stream>>>(
        Hb, WPRJ + (size_t)l * 256 * 256, proj_b + l * Dm, X, X, (int)ROWS, 256);
    ln_kernel<0, true><<<(int)(ROWS / 8), 256, 0, stream>>>(
        X, ln2_g + l * Dm, ln2_b + l * Dm, Hb, (int)ROWS);
    gemm_kernel<256, false, true, 1><<<dim3(Ttok, 4), 256, 0, stream>>>(
        Hb, WFC1 + (size_t)l * 512 * 256, fc1_b + l * MLPD, nullptr, Fb,
        (int)ROWS, 512);
    gemm_kernel<512, true, false, 0><<<dim3(Ttok, 2), 256, 0, stream>>>(
        Fb, WFC2 + (size_t)l * 256 * 512, fc2_b + l * Dm, X, X, (int)ROWS, 256);
  }

  // final norm (f32 out; Y overlaps the now-dead qkv buffers)
  ln_kernel<0, false><<<(int)(ROWS / 8), 256, 0, stream>>>(X, norm_g, norm_b, Y,
                                                           (int)ROWS);

  // policy head: rows = B*576 (reference slices x[:, 1:N+1])
  const int PROWS = BSZ * Ntok;  // 9216
  ln_kernel<1, true><<<PROWS / 8, 256, 0, stream>>>(Y, pol_ln_g, pol_ln_b, Hb, PROWS);
  gemm_kernel<256, false, true, 1><<<dim3(PROWS / 16, 2), 256, 0, stream>>>(
      Hb, WPOL, pol_b1, nullptr, Fb, PROWS, 256);
  head_dot_kernel<false><<<PROWS / 8, 256, 0, stream>>>(Fb, pol_w2, pol_b2, out, PROWS);

  // value head: rows = B
  ln_kernel<2, true><<<2, 256, 0, stream>>>(Y, val_ln_g, val_ln_b, VHb, BSZ);
  gemm_kernel<256, false, true, 1><<<dim3(1, 2), 256, 0, stream>>>(
      VHb, WVAL, val_b1, nullptr, VFb, BSZ, 256);
  head_dot_kernel<true><<<2, 256, 0, stream>>>(VFb, val_w2, val_b2, out, BSZ);
}
